// QAttn_32598801776759
// MI455X (gfx1250) — matrix operand order repaired and checked by reading, not compiled, not run on hardware
//
#include <hip/hip_runtime.h>

typedef __attribute__((ext_vector_type(16))) _Float16 v16h;
typedef __attribute__((ext_vector_type(8)))  _Float16 v8h;
typedef __attribute__((ext_vector_type(8)))  float    v8f;
typedef __attribute__((ext_vector_type(4)))  unsigned int u32x4;
typedef __attribute__((ext_vector_type(8)))  int      i32x8;
typedef __attribute__((ext_vector_type(4)))  int      i32x4;

#define BB 2
#define NN 2048
#define CC 1024
#define HH 16
#define DD 64

// ---- Tensor Data Mover availability (builtin arity differs by toolchain) ----
#if defined(__has_builtin)
#if __has_builtin(__builtin_amdgcn_tensor_load_to_lds) && \
    __has_builtin(__builtin_amdgcn_s_wait_tensorcnt)
#define HAVE_TDM 1
#endif
#endif
#ifndef HAVE_TDM
#define HAVE_TDM 0
#endif

static __device__ __forceinline__ v16h combine16(v8h lo, v8h hi) {
    v16h r;
#pragma unroll
    for (int i = 0; i < 8; ++i) { r[i] = lo[i]; r[8 + i] = hi[i]; }
    return r;
}

#if HAVE_TDM
// Issue one 2-D TDM tile load (f16 elements) per CDNA5 ISA D# layout
// (CDNA5 ISA).  All args are wave-uniform.
static __device__ __forceinline__ void tdm_load_tile_2d(
    const void* lds_dst, const void* gsrc,
    unsigned tensor_d0, unsigned tensor_d1, unsigned stride0_elems,
    unsigned tile_d0, unsigned tile_d1)
{
    unsigned long long ga = (unsigned long long)gsrc;
    unsigned lds = (unsigned)(unsigned long long)lds_dst;   // LDS byte address
    u32x4 g0;
    g0[0] = 1u;                                       // count=1 (valid user D#)
    g0[1] = lds;                                      // lds_addr
    g0[2] = (unsigned)ga;                             // global_addr[31:0]
    g0[3] = (unsigned)((ga >> 32) & 0x01FFFFFFull)    // global_addr[56:32]
          | (2u << 30);                               // type = 2 ("image")
    i32x8 g1;
    g1[0] = (int)(1u << 16);                          // wg_mask=0, data_size=1 (2B)
    g1[1] = (int)((tensor_d0 & 0xFFFFu) << 16);       // tensor_dim0[15:0]
    g1[2] = (int)((tensor_d0 >> 16) | ((tensor_d1 & 0xFFFFu) << 16));
    g1[3] = (int)((tensor_d1 >> 16) | (tile_d0 << 16));  // tile_dim0
    g1[4] = (int)(tile_d1 & 0xFFFFu);                 // tile_dim1; tile_dim2=0
    g1[5] = (int)stride0_elems;                       // tensor_dim0_stride[31:0]
    g1[6] = 0;                                        // stride0 hi / stride1 lo
    g1[7] = 0;
    i32x4 z4 = {0, 0, 0, 0};
#if __clang_major__ >= 23
    i32x8 z8 = {0, 0, 0, 0, 0, 0, 0, 0};
    __builtin_amdgcn_tensor_load_to_lds(g0, g1, z4, z4, z8, 0);
#else
    __builtin_amdgcn_tensor_load_to_lds(g0, g1, z4, z4, 0);
#endif
}
#endif

// ---------------------------------------------------------------------------
// Kernel 1: fused Q/K/V projections.  out = x @ W  (M=4096, K=1024, N=1024)
// Block tile 128(M) x 64(N), 8 waves, each wave 32x32 (2x2 wmma tiles).
// Q,K stored f16 as [b][h][n][d]; V stored transposed [b][h][d][n].
// ---------------------------------------------------------------------------
__global__ __launch_bounds__(256)
void proj_qkv_kernel(const float* __restrict__ x,
                     const float* __restrict__ Wq,
                     const float* __restrict__ Wk,
                     const float* __restrict__ Wv,
                     _Float16* __restrict__ q_ws,
                     _Float16* __restrict__ k_ws,
                     _Float16* __restrict__ vT_ws)
{
    __shared__ __align__(16) _Float16 lds_a[128 * 32];   // A tile [m][k]
    __shared__ __align__(16) _Float16 lds_bT[64 * 32];   // B tile transposed [n][k]

    const int mode = blockIdx.z;  // 0:Q 1:K 2:V
    const float* __restrict__ W = (mode == 0) ? Wq : (mode == 1) ? Wk : Wv;

    const int m_base = blockIdx.x * 128;
    const int n_base = blockIdx.y * 64;

    const int t    = threadIdx.x;
    const int lane = t & 31;
    const int wave = t >> 5;
    const int l    = lane & 15;
    const int h    = lane >> 4;
    const int wm   = (wave >> 1) * 32;
    const int wn   = (wave & 1) * 32;

    v8f acc[2][2] = {};

    for (int k0 = 0; k0 < CC; k0 += 32) {
        // Stage A: 128x32 fp32 -> f16 (coalesced global reads)
        {
            int r  = t >> 1;
            int c0 = (t & 1) * 16;
            const float* src = x + (size_t)(m_base + r) * CC + k0 + c0;
            if (k0 + 32 < CC) __builtin_prefetch(src + 32, 0, 3);
            v8h o0, o1;
#pragma unroll
            for (int i = 0; i < 8; ++i) { o0[i] = (_Float16)src[i]; o1[i] = (_Float16)src[8 + i]; }
            *(v8h*)&lds_a[r * 32 + c0]     = o0;
            *(v8h*)&lds_a[r * 32 + c0 + 8] = o1;
        }
        // Stage B transposed: W[k0..+31][n_base..+63] -> lds_bT[n][k]
#pragma unroll
        for (int i = 0; i < 2; ++i) {
            int f  = t * 2 + i;
            int kr = f >> 4;
            int nc = (f & 15) * 4;
            const float* src = W + (size_t)(k0 + kr) * CC + n_base + nc;
            if (i == 0 && k0 + 32 < CC) __builtin_prefetch(src + 32 * CC, 0, 3);
#pragma unroll
            for (int j = 0; j < 4; ++j)
                lds_bT[(nc + j) * 32 + kr] = (_Float16)src[j];
        }
        __syncthreads();

        v16h afrag[2], bfrag[2];
#pragma unroll
        for (int mt = 0; mt < 2; ++mt) {
            int row = wm + mt * 16 + l;
            afrag[mt] = combine16(*(const v8h*)&lds_a[row * 32 + h * 8],
                                  *(const v8h*)&lds_a[row * 32 + 16 + h * 8]);
        }
#pragma unroll
        for (int nt = 0; nt < 2; ++nt) {
            int col = wn + nt * 16 + l;
            bfrag[nt] = combine16(*(const v8h*)&lds_bT[col * 32 + h * 8],
                                  *(const v8h*)&lds_bT[col * 32 + 16 + h * 8]);
        }
#pragma unroll
        for (int mt = 0; mt < 2; ++mt)
#pragma unroll
            for (int nt = 0; nt < 2; ++nt)
                acc[mt][nt] = __builtin_amdgcn_wmma_f32_16x16x32_f16(
                    false, afrag[mt], false, bfrag[nt], (short)0, acc[mt][nt], false, false);
        __syncthreads();
    }

    // Block-uniform decomposition: 128-row tile stays in one batch, 64-col
    // tile stays in one head -> b, hh, bh all uniform (no per-element branches).
    const int b       = m_base >> 11;
    const int tokbase = (m_base & (NN - 1)) + wm;
    const int bh      = b * HH + (n_base >> 6);

    if (mode == 2) {
        // V^T layout [bh][d][tok]: contiguous in v -> vectorizable stores.
#pragma unroll
        for (int mt = 0; mt < 2; ++mt)
#pragma unroll
            for (int nt = 0; nt < 2; ++nt) {
                int d = wn + nt * 16 + l;
                _Float16* dst = vT_ws + ((size_t)bh * DD + d) * NN
                              + tokbase + mt * 16 + h * 8;
#pragma unroll
                for (int v = 0; v < 8; ++v)
                    dst[v] = (_Float16)acc[mt][nt][v];
            }
    } else {
        _Float16* base = ((mode == 0) ? q_ws : k_ws) + (size_t)bh * NN * DD;
#pragma unroll
        for (int mt = 0; mt < 2; ++mt)
#pragma unroll
            for (int nt = 0; nt < 2; ++nt) {
                int d = wn + nt * 16 + l;
#pragma unroll
                for (int v = 0; v < 8; ++v)
                    base[(size_t)(tokbase + mt * 16 + v + h * 8) * DD + d] =
                        (_Float16)acc[mt][nt][v];
            }
    }
}

// ---------------------------------------------------------------------------
// Kernel 2: flash attention per (b,h).  4 waves x 16 query rows = 64-row tile.
// K/V^T tiles staged into double-buffered LDS by the Tensor Data Mover
// (wave 0 issues 2 descriptors/tile; s_wait_tensorcnt<=2 retires the oldest
// tile while the next one streams in), overlapping DMA with WMMA + softmax.
// ---------------------------------------------------------------------------
__global__ __launch_bounds__(128)
void flash_attn_kernel(const _Float16* __restrict__ q_ws,
                       const _Float16* __restrict__ k_ws,
                       const _Float16* __restrict__ vT_ws,
                       _Float16* __restrict__ attn_ws)
{
    __shared__ __align__(16) _Float16 lds_k[2][32 * 64];  // K tile [j][d]   (B^T for QK^T)
    __shared__ __align__(16) _Float16 lds_v[2][64 * 32];  // V^T tile [d][j] (B^T for PV)
    __shared__ __align__(16) _Float16 lds_p[64 * 32];     // P tile [m][j]

    const int bh    = blockIdx.y;
    const int qbase = blockIdx.x * 64;
    const int t     = threadIdx.x;
    const int lane  = t & 31;
    const int wave  = t >> 5;
    const int l     = lane & 15;
    const int h     = lane >> 4;

    const _Float16* __restrict__ Qh = q_ws + (size_t)bh * NN * DD;
    const _Float16* __restrict__ Kh = k_ws + (size_t)bh * NN * DD;
    const _Float16* __restrict__ Vt = vT_ws + (size_t)bh * DD * NN;

    // Q A-fragments held in registers for the whole key loop.
    v16h qf[2];
    {
        const _Float16* qr = Qh + (size_t)(qbase + wave * 16 + l) * DD;
#pragma unroll
        for (int kk = 0; kk < 2; ++kk)
            qf[kk] = combine16(*(const v8h*)(qr + kk * 32 + h * 8),
                               *(const v8h*)(qr + kk * 32 + 16 + h * 8));
    }

    v8f   o_acc[4] = {};
    float mrow[8], lrow[8];
#pragma unroll
    for (int v = 0; v < 8; ++v) { mrow[v] = -3.0e38f; lrow[v] = 0.0f; }

    const int nIter = NN / 32;

#if HAVE_TDM
    if (wave == 0) {
        tdm_load_tile_2d(&lds_k[0][0], Kh, DD, NN, DD, DD, 32);      // 32x64 row tile
        tdm_load_tile_2d(&lds_v[0][0], Vt, NN, DD, NN, 32, DD);      // 64x32 strided tile
    }
#endif

    for (int it = 0; it < nIter; ++it) {
        const int j0  = it * 32;
        const int cur = it & 1;

#if HAVE_TDM
        if (wave == 0) {
            if (it + 1 < nIter) {
                const int j1 = j0 + 32;
                tdm_load_tile_2d(&lds_k[cur ^ 1][0], Kh + (size_t)j1 * DD,
                                 DD, NN, DD, DD, 32);
                tdm_load_tile_2d(&lds_v[cur ^ 1][0], Vt + j1,
                                 NN, DD, NN, 32, DD);
                __builtin_amdgcn_s_wait_tensorcnt(2);  // tile `it` (oldest 2) done
            } else {
                __builtin_amdgcn_s_wait_tensorcnt(0);
            }
        }
        __syncthreads();
#else
        // Fallback: synchronous vector staging.
        {
            int r  = t >> 2;
            int c0 = (t & 3) * 16;
            const v8h* src = (const v8h*)(Kh + (size_t)(j0 + r) * DD + c0);
            *(v8h*)&lds_k[cur][r * 64 + c0]     = src[0];
            *(v8h*)&lds_k[cur][r * 64 + c0 + 8] = src[1];
        }
        {
            int r  = t >> 1;
            int c0 = (t & 1) * 16;
            const v8h* src = (const v8h*)(Vt + (size_t)r * NN + j0 + c0);
            *(v8h*)&lds_v[cur][r * 32 + c0]     = src[0];
            *(v8h*)&lds_v[cur][r * 32 + c0 + 8] = src[1];
        }
        __syncthreads();
#endif

        const _Float16* __restrict__ Kt = lds_k[cur];
        const _Float16* __restrict__ Vl = lds_v[cur];

        // S = Q K^T  (16 rows x 32 key cols per wave)
        v8f s[2];
#pragma unroll
        for (int nt = 0; nt < 2; ++nt) {
            v8f a = {};
            int col = nt * 16 + l;
#pragma unroll
            for (int kk = 0; kk < 2; ++kk) {
                v16h bf = combine16(*(const v8h*)&Kt[col * 64 + kk * 32 + h * 8],
                                    *(const v8h*)&Kt[col * 64 + kk * 32 + 16 + h * 8]);
                a = __builtin_amdgcn_wmma_f32_16x16x32_f16(
                        false, qf[kk], false, bf, (short)0, a, false, false);
            }
            s[nt] = a;
        }

        // Online softmax: per-row butterfly reduction inside each 16-lane half.
#pragma unroll
        for (int v = 0; v < 8; ++v) {
            float s0 = s[0][v] * 0.125f;   // 1/sqrt(64)
            float s1 = s[1][v] * 0.125f;
            float mx = fmaxf(s0, s1);
#pragma unroll
            for (int msk = 1; msk <= 8; msk <<= 1) mx = fmaxf(mx, __shfl_xor(mx, msk, 32));
            float mnew = fmaxf(mrow[v], mx);
            float p0 = __expf(s0 - mnew);
            float p1 = __expf(s1 - mnew);
            float corr = __expf(mrow[v] - mnew);
            float sm = p0 + p1;
#pragma unroll
            for (int msk = 1; msk <= 8; msk <<= 1) sm += __shfl_xor(sm, msk, 32);
            lrow[v] = lrow[v] * corr + sm;
            mrow[v] = mnew;
#pragma unroll
            for (int dt = 0; dt < 4; ++dt) o_acc[dt][v] *= corr;
            // C-layout -> A-layout transpose of P via per-wave LDS rows.
            int m_loc = wave * 16 + v + h * 8;
            lds_p[m_loc * 32 + l]      = (_Float16)p0;
            lds_p[m_loc * 32 + 16 + l] = (_Float16)p1;
        }

        // O += P @ V (same-wave LDS RAW covered by compiler dscnt waits)
        v16h pf;
        {
            int row = wave * 16 + l;
            pf = combine16(*(const v8h*)&lds_p[row * 32 + h * 8],
                           *(const v8h*)&lds_p[row * 32 + 16 + h * 8]);
        }
#pragma unroll
        for (int dt = 0; dt < 4; ++dt) {
            int d = dt * 16 + l;
            v16h vf = combine16(*(const v8h*)&Vl[d * 32 + h * 8],
                                *(const v8h*)&Vl[d * 32 + 16 + h * 8]);
            o_acc[dt] = __builtin_amdgcn_wmma_f32_16x16x32_f16(
                            false, pf, false, vf, (short)0, o_acc[dt], false, false);
        }
        __syncthreads();
    }

    // Normalize and store to concat layout [b][n][h*64+d] (f16).
    const int b = bh >> 4, hh = bh & 15;
#pragma unroll
    for (int dt = 0; dt < 4; ++dt) {
#pragma unroll
        for (int v = 0; v < 8; ++v) {
            int row = qbase + wave * 16 + v + h * 8;
            float val = o_acc[dt][v] / lrow[v];
            attn_ws[((size_t)(b * NN + row)) * CC + hh * 64 + dt * 16 + l] = (_Float16)val;
        }
    }
}

// ---------------------------------------------------------------------------
// Kernel 3: output projection.  out(f32) = concat(f16) @ W_o(f32).
// ---------------------------------------------------------------------------
__global__ __launch_bounds__(256)
void out_proj_kernel(const _Float16* __restrict__ a_in,
                     const float* __restrict__ Wo,
                     float* __restrict__ out)
{
    __shared__ __align__(16) _Float16 lds_a[128 * 32];
    __shared__ __align__(16) _Float16 lds_bT[64 * 32];

    const int m_base = blockIdx.x * 128;
    const int n_base = blockIdx.y * 64;

    const int t    = threadIdx.x;
    const int lane = t & 31;
    const int wave = t >> 5;
    const int l    = lane & 15;
    const int h    = lane >> 4;
    const int wm   = (wave >> 1) * 32;
    const int wn   = (wave & 1) * 32;

    v8f acc[2][2] = {};

    for (int k0 = 0; k0 < CC; k0 += 32) {
        {
            int r  = t >> 1;
            int c0 = (t & 1) * 16;
            const v8h* src = (const v8h*)(a_in + (size_t)(m_base + r) * CC + k0 + c0);
            if (k0 + 32 < CC) __builtin_prefetch((const _Float16*)src + 32, 0, 3);
            *(v8h*)&lds_a[r * 32 + c0]     = src[0];
            *(v8h*)&lds_a[r * 32 + c0 + 8] = src[1];
        }
#pragma unroll
        for (int i = 0; i < 2; ++i) {
            int f  = t * 2 + i;
            int kr = f >> 4;
            int nc = (f & 15) * 4;
            const float* src = Wo + (size_t)(k0 + kr) * CC + n_base + nc;
            if (i == 0 && k0 + 32 < CC) __builtin_prefetch(src + 32 * CC, 0, 3);
#pragma unroll
            for (int j = 0; j < 4; ++j)
                lds_bT[(nc + j) * 32 + kr] = (_Float16)src[j];
        }
        __syncthreads();

        v16h afrag[2], bfrag[2];
#pragma unroll
        for (int mt = 0; mt < 2; ++mt) {
            int row = wm + mt * 16 + l;
            afrag[mt] = combine16(*(const v8h*)&lds_a[row * 32 + h * 8],
                                  *(const v8h*)&lds_a[row * 32 + 16 + h * 8]);
        }
#pragma unroll
        for (int nt = 0; nt < 2; ++nt) {
            int col = wn + nt * 16 + l;
            bfrag[nt] = combine16(*(const v8h*)&lds_bT[col * 32 + h * 8],
                                  *(const v8h*)&lds_bT[col * 32 + 16 + h * 8]);
        }
#pragma unroll
        for (int mt = 0; mt < 2; ++mt)
#pragma unroll
            for (int nt = 0; nt < 2; ++nt)
                acc[mt][nt] = __builtin_amdgcn_wmma_f32_16x16x32_f16(
                    false, afrag[mt], false, bfrag[nt], (short)0, acc[mt][nt], false, false);
        __syncthreads();
    }

#pragma unroll
    for (int mt = 0; mt < 2; ++mt)
#pragma unroll
        for (int nt = 0; nt < 2; ++nt) {
            int col = n_base + wn + nt * 16 + l;
#pragma unroll
            for (int v = 0; v < 8; ++v) {
                int m = m_base + wm + mt * 16 + v + h * 8;
                out[(size_t)m * CC + col] = acc[mt][nt][v];
            }
        }
}

// ---------------------------------------------------------------------------
extern "C" void kernel_launch(void* const* d_in, const int* in_sizes, int n_in,
                              void* d_out, int out_size, void* d_ws, size_t ws_size,
                              hipStream_t stream)
{
    const float* x  = (const float*)d_in[0];
    const float* Wq = (const float*)d_in[1];
    const float* Wk = (const float*)d_in[2];
    const float* Wv = (const float*)d_in[3];
    const float* Wo = (const float*)d_in[4];
    float* out = (float*)d_out;

    const size_t per = (size_t)BB * HH * NN * DD;  // 4M halves each
    _Float16* q_ws   = (_Float16*)d_ws;
    _Float16* k_ws   = q_ws + per;
    _Float16* vT_ws  = k_ws + per;
    _Float16* att_ws = vT_ws + per;

    proj_qkv_kernel<<<dim3((BB * NN) / 128, CC / 64, 3), 256, 0, stream>>>(
        x, Wq, Wk, Wv, q_ws, k_ws, vT_ws);

    flash_attn_kernel<<<dim3(NN / 64, BB * HH), 128, 0, stream>>>(
        q_ws, k_ws, vT_ws, att_ws);

    out_proj_kernel<<<dim3((BB * NN) / 128, CC / 64), 256, 0, stream>>>(
        att_ws, Wo, out);
}
